// MultiHeadAttention_78116865179844
// MI455X (gfx1250) — matrix operand order repaired and checked by reading, not compiled, not run on hardware
//
#include <hip/hip_runtime.h>
#include <hip/hip_bf16.h>
#include <math.h>

// ---------------------------------------------------------------------------
// MHA block for MI455X (gfx1250, wave32, WMMA).
// fp32 emulated by bf16 hi/lo split (3-term WMMA). All matrix operands are
// pre-packed into bf16 hi/lo planes so the WMMA inner loops carry no
// f32->bf16 conversion VALU work.
// ---------------------------------------------------------------------------

#define NHEADS 12
#define BATCH  4
#define SEQ    1024
#define DMODEL 768
#define DHEAD  64
#define LENREG 16
#define TCORE  (SEQ - 1 - LENREG)   // 1007 core tokens (RoPE'd)

typedef __attribute__((ext_vector_type(16))) __bf16 v16bf;
typedef __attribute__((ext_vector_type(8)))  float  v8f;

__device__ __forceinline__ void split_bf(float x, __bf16& hi, __bf16& lo) {
    hi = (__bf16)x;
    lo = (__bf16)(x - (float)hi);
}

__device__ __forceinline__ v8f wmma_bf16(v16bf a, v16bf b, v8f c) {
    // D = A(16x32 bf16) * B(32x16 bf16) + C(16x16 f32)
    return __builtin_amdgcn_wmma_f32_16x16x32_bf16(false, a, false, b,
                                                   (short)0, c, false, false);
}

// A-fragment (16x32, 16-bit): lane m = lane&15, kb8 = (lane<16)?0:8.
// element e -> K = 2*kb8 + e (contiguous, matches B fragment)
__device__ __forceinline__ void load_a_frag_bf(const __bf16* hi_row,
                                               const __bf16* lo_row, int kb8,
                                               v16bf& hi, v16bf& lo) {
#pragma unroll
    for (int e = 0; e < 16; ++e) {
        const int k = 2 * kb8 + e;
        hi[e] = hi_row[k];
        lo[e] = lo_row[k];
    }
}

// f32 variant (used only for the P tile inside attention)
__device__ __forceinline__ void load_a_frag_f32(const float* rowptr, int kb8,
                                                v16bf& hi, v16bf& lo) {
#pragma unroll
    for (int e = 0; e < 16; ++e) {
        const int k = 2 * kb8 + e;
        __bf16 h, l; split_bf(rowptr[k], h, l);
        hi[e] = h; lo[e] = l;
    }
}

// B-fragment (32x16, 16-bit): lane n = lane&15, kb16 = (lane<16)?0:16,
// element e -> K = kb16 + e. Pointers pre-offset to W[n] + k0 + kb16,
// 32B-aligned by construction -> single v16bf vector load each.
__device__ __forceinline__ void load_b_frag_bf(const __bf16* hi_p,
                                               const __bf16* lo_p,
                                               v16bf& hi, v16bf& lo) {
    hi = *(const v16bf*)hi_p;
    lo = *(const v16bf*)lo_p;
}

// ---------------------------------------------------------------------------
// Kernel 0: elementwise f32 -> (bf16 hi, bf16 lo) pack.
// ---------------------------------------------------------------------------
__global__ void pack_kernel(const float* __restrict__ src,
                            __bf16* __restrict__ hi, __bf16* __restrict__ lo,
                            int n) {
    const int i = blockIdx.x * blockDim.x + threadIdx.x;
    if (i >= n) return;
    __bf16 h, l; split_bf(src[i], h, l);
    hi[i] = h; lo[i] = l;
}

// ---------------------------------------------------------------------------
// Kernel 1: QKV projection. C(4096 x 2304) = x(4096 x 768) @ w_qkv^T + b.
// One wave computes a 16(M) x 64(N) tile from packed operands and scatters
// directly into packed Q/K/V hi/lo planes, layout (B,H,N,Dh).
// ---------------------------------------------------------------------------
__global__ __launch_bounds__(32)
void qkv_gemm_kernel(const __bf16* __restrict__ xhi, const __bf16* __restrict__ xlo,
                     const __bf16* __restrict__ whi, const __bf16* __restrict__ wlo,
                     const float* __restrict__ bias,
                     __bf16* __restrict__ qhi, __bf16* __restrict__ qlo,
                     __bf16* __restrict__ khi, __bf16* __restrict__ klo,
                     __bf16* __restrict__ vhi, __bf16* __restrict__ vlo) {
    const int lane = threadIdx.x;
    const int ln   = lane & 15;
    const int half = lane >> 4;
    const int kb8  = half ? 8 : 0;
    const int kb16 = half ? 16 : 0;
    const int tn   = blockIdx.x;             // 64-col tile: 0..35
    const int tm   = blockIdx.y;             // 16-row tile: 0..255
    const size_t arow = (size_t)(tm * 16 + ln) * DMODEL;

    v8f c[4];
    const v8f zero = {0,0,0,0,0,0,0,0};
#pragma unroll
    for (int t = 0; t < 4; ++t) c[t] = zero;

    for (int k0 = 0; k0 < DMODEL; k0 += 32) {
        v16bf ah, al;
        load_a_frag_bf(xhi + arow + k0, xlo + arow + k0, kb8, ah, al);
#pragma unroll
        for (int t = 0; t < 4; ++t) {
            const size_t woff = (size_t)(tn * 64 + t * 16 + ln) * DMODEL + k0 + kb16;
            v16bf bh, bl;
            load_b_frag_bf(whi + woff, wlo + woff, bh, bl);
            c[t] = wmma_bf16(ah, bh, c[t]);
            c[t] = wmma_bf16(ah, bl, c[t]);
            c[t] = wmma_bf16(al, bh, c[t]);
        }
    }

#pragma unroll
    for (int t = 0; t < 4; ++t) {
        const int ncol  = tn * 64 + t * 16 + ln;
        const float bv  = bias[ncol];
        const int which = ncol / DMODEL;        // 0=q 1=k 2=v
        const int cm    = ncol % DMODEL;
        const int h     = cm / DHEAD;
        const int d     = cm % DHEAD;
        __bf16* dhi = (which == 0) ? qhi : ((which == 1) ? khi : vhi);
        __bf16* dlo = (which == 0) ? qlo : ((which == 1) ? klo : vlo);
#pragma unroll
        for (int r = 0; r < 8; ++r) {
            const int m    = tm * 16 + r + 8 * half;   // C layout: row r+8*(lane>>4)
            const int b    = m / SEQ;
            const int ntok = m % SEQ;
            const size_t o = (((size_t)b * NHEADS + h) * SEQ + ntok) * DHEAD + d;
            __bf16 hh, ll; split_bf(c[t][r] + bv, hh, ll);
            dhi[o] = hh; dlo[o] = ll;
        }
    }
}

// ---------------------------------------------------------------------------
// Kernel 2: RoPE on core tokens [1, 1+TCORE) of packed Q and K, in place.
// f32 value reconstructed as hi+lo, rotated, re-split.
// ---------------------------------------------------------------------------
__global__ void rope_kernel(__bf16* __restrict__ qhi, __bf16* __restrict__ qlo,
                            __bf16* __restrict__ khi, __bf16* __restrict__ klo,
                            const int* __restrict__ position,
                            const float* __restrict__ freqs) {
    const int idx   = blockIdx.x * blockDim.x + threadIdx.x;
    const int total = BATCH * NHEADS * TCORE * (DHEAD / 2);
    if (idx >= total) return;
    const int i = idx & 31;                       // freq index (DHEAD/2 == 32)
    const int t = (idx >> 5) % TCORE;
    const int h = (idx / (32 * TCORE)) % NHEADS;
    const int b = idx / (32 * TCORE * NHEADS);

    const float ang = (float)position[b * TCORE + t] * freqs[i];
    const float ca = cosf(ang), sa = sinf(ang);
    const size_t base = (((size_t)b * NHEADS + h) * SEQ + (1 + t)) * DHEAD + 2 * i;

    {
        float q0 = (float)qhi[base]     + (float)qlo[base];
        float q1 = (float)qhi[base + 1] + (float)qlo[base + 1];
        __bf16 h0, l0, h1, l1;
        split_bf(q0 * ca - q1 * sa, h0, l0);
        split_bf(q0 * sa + q1 * ca, h1, l1);
        qhi[base] = h0; qlo[base] = l0;
        qhi[base + 1] = h1; qlo[base + 1] = l1;
    }
    {
        float k0 = (float)khi[base]     + (float)klo[base];
        float k1 = (float)khi[base + 1] + (float)klo[base + 1];
        __bf16 h0, l0, h1, l1;
        split_bf(k0 * ca - k1 * sa, h0, l0);
        split_bf(k0 * sa + k1 * ca, h1, l1);
        khi[base] = h0; klo[base] = l0;
        khi[base + 1] = h1; klo[base + 1] = l1;
    }
}

// ---------------------------------------------------------------------------
// Kernel 3: flash attention. One wave per (b, h, 16-query tile).
// Online softmax; P transposed through LDS into A-fragment layout.
// Output written already packed for the projection GEMM.
// ---------------------------------------------------------------------------
__global__ __launch_bounds__(32)
void attn_kernel(const __bf16* __restrict__ qhi, const __bf16* __restrict__ qlo,
                 const __bf16* __restrict__ khi, const __bf16* __restrict__ klo,
                 const __bf16* __restrict__ vhi, const __bf16* __restrict__ vlo,
                 __bf16* __restrict__ ohi, __bf16* __restrict__ olo) {
    __shared__ float plds[16][32];
    const int lane = threadIdx.x;
    const int ln   = lane & 15;
    const int half = lane >> 4;
    const int kb8  = half ? 8 : 0;
    const int kb16 = half ? 16 : 0;
    const int qt = blockIdx.x;               // 0..63
    const int h  = blockIdx.y;               // 0..11
    const int b  = blockIdx.z;               // 0..3

    const size_t bh   = (size_t)b * NHEADS + h;
    const size_t base = bh * SEQ * DHEAD;

    // Q fragments, K-dim 64 in two 32-steps (loaded once, already packed)
    v16bf qfh[2], qfl[2];
#pragma unroll
    for (int s = 0; s < 2; ++s) {
        const size_t qoff = base + (size_t)(qt * 16 + ln) * DHEAD + s * 32;
        load_a_frag_bf(qhi + qoff, qlo + qoff, kb8, qfh[s], qfl[s]);
    }

    v8f o[4];
    const v8f zero = {0,0,0,0,0,0,0,0};
#pragma unroll
    for (int t = 0; t < 4; ++t) o[t] = zero;
    float mrun[8], lrun[8];
#pragma unroll
    for (int r = 0; r < 8; ++r) { mrun[r] = -1e30f; lrun[r] = 0.0f; }

    for (int j = 0; j < SEQ; j += 32) {
        // ---- scores: two 16x16 tiles (keys j..j+15, j+16..j+31) ----
        v8f s0 = zero, s1 = zero;
#pragma unroll
        for (int s = 0; s < 2; ++s) {
            const size_t k0off = base + (size_t)(j + ln)      * DHEAD + s * 32 + kb16;
            const size_t k1off = base + (size_t)(j + 16 + ln) * DHEAD + s * 32 + kb16;
            v16bf b0h, b0l, b1h, b1l;
            load_b_frag_bf(khi + k0off, klo + k0off, b0h, b0l);
            load_b_frag_bf(khi + k1off, klo + k1off, b1h, b1l);
            s0 = wmma_bf16(qfh[s], b0h, s0);
            s0 = wmma_bf16(qfh[s], b0l, s0);
            s0 = wmma_bf16(qfl[s], b0h, s0);
            s1 = wmma_bf16(qfh[s], b1h, s1);
            s1 = wmma_bf16(qfh[s], b1l, s1);
            s1 = wmma_bf16(qfl[s], b1h, s1);
        }

        // ---- online softmax (per row r + 8*half) ----
        float corr[8];
#pragma unroll
        for (int r = 0; r < 8; ++r) {
            float a0 = s0[r] * 0.125f;      // 1/sqrt(64)
            float a1 = s1[r] * 0.125f;
            float mx = fmaxf(a0, a1);
            mx = fmaxf(mx, __shfl_xor(mx, 1, 32));
            mx = fmaxf(mx, __shfl_xor(mx, 2, 32));
            mx = fmaxf(mx, __shfl_xor(mx, 4, 32));
            mx = fmaxf(mx, __shfl_xor(mx, 8, 32));
            const float mnew = fmaxf(mrun[r], mx);
            corr[r] = __expf(mrun[r] - mnew);
            mrun[r] = mnew;
            const float p0 = __expf(a0 - mnew);
            const float p1 = __expf(a1 - mnew);
            s0[r] = p0; s1[r] = p1;
            float rs = p0 + p1;
            rs += __shfl_xor(rs, 1, 32);
            rs += __shfl_xor(rs, 2, 32);
            rs += __shfl_xor(rs, 4, 32);
            rs += __shfl_xor(rs, 8, 32);
            lrun[r] = lrun[r] * corr[r] + rs;
        }
#pragma unroll
        for (int t = 0; t < 4; ++t)
#pragma unroll
            for (int r = 0; r < 8; ++r) o[t][r] *= corr[r];

        // ---- transpose P (C layout -> A layout) through LDS ----
        __syncthreads();
#pragma unroll
        for (int r = 0; r < 8; ++r) {
            plds[r + 8 * half][ln]      = s0[r];
            plds[r + 8 * half][ln + 16] = s1[r];
        }
        __syncthreads();
        v16bf ph, pl;
        load_a_frag_f32(&plds[ln][0], kb8, ph, pl);

        // ---- O += P @ V (four 16-wide d tiles), V already packed ----
#pragma unroll
        for (int t = 0; t < 4; ++t) {
            v16bf vfh, vfl;
#pragma unroll
            for (int e = 0; e < 16; ++e) {
                const size_t voff = base + (size_t)(j + kb16 + e) * DHEAD + t * 16 + ln;
                vfh[e] = vhi[voff];
                vfl[e] = vlo[voff];
            }
            o[t] = wmma_bf16(ph, vfh, o[t]);
            o[t] = wmma_bf16(ph, vfl, o[t]);
            o[t] = wmma_bf16(pl, vfh, o[t]);
        }
        __syncthreads();
    }

    // ---- normalize and store packed as (B, N, H, Dh) ----
#pragma unroll
    for (int t = 0; t < 4; ++t)
#pragma unroll
        for (int r = 0; r < 8; ++r) {
            const int ntok = qt * 16 + r + 8 * half;
            const size_t oo = (((size_t)b * SEQ + ntok) * NHEADS + h) * DHEAD + t * 16 + ln;
            __bf16 hh, ll; split_bf(o[t][r] / lrun[r], hh, ll);
            ohi[oo] = hh; olo[oo] = ll;
        }
}

// ---------------------------------------------------------------------------
// Kernel 4: output projection. out(4096x768) = attn(4096x768) @ w_fc^T + b_fc.
// ---------------------------------------------------------------------------
__global__ __launch_bounds__(32)
void proj_gemm_kernel(const __bf16* __restrict__ ahi, const __bf16* __restrict__ alo,
                      const __bf16* __restrict__ whi, const __bf16* __restrict__ wlo,
                      const float* __restrict__ bias, float* __restrict__ out) {
    const int lane = threadIdx.x;
    const int ln   = lane & 15;
    const int half = lane >> 4;
    const int kb8  = half ? 8 : 0;
    const int kb16 = half ? 16 : 0;
    const int tn   = blockIdx.x;            // 0..11 (64-col tiles)
    const int tm   = blockIdx.y;            // 0..255
    const size_t arow = (size_t)(tm * 16 + ln) * DMODEL;

    v8f c[4];
    const v8f zero = {0,0,0,0,0,0,0,0};
#pragma unroll
    for (int t = 0; t < 4; ++t) c[t] = zero;

    for (int k0 = 0; k0 < DMODEL; k0 += 32) {
        v16bf ah, al;
        load_a_frag_bf(ahi + arow + k0, alo + arow + k0, kb8, ah, al);
#pragma unroll
        for (int t = 0; t < 4; ++t) {
            const size_t woff = (size_t)(tn * 64 + t * 16 + ln) * DMODEL + k0 + kb16;
            v16bf bh, bl;
            load_b_frag_bf(whi + woff, wlo + woff, bh, bl);
            c[t] = wmma_bf16(ah, bh, c[t]);
            c[t] = wmma_bf16(ah, bl, c[t]);
            c[t] = wmma_bf16(al, bh, c[t]);
        }
    }
#pragma unroll
    for (int t = 0; t < 4; ++t) {
        const int ncol = tn * 64 + t * 16 + ln;
        const float bv = bias[ncol];
#pragma unroll
        for (int r = 0; r < 8; ++r) {
            const int m = tm * 16 + r + 8 * half;
            out[(size_t)m * DMODEL + ncol] = c[t][r] + bv;
        }
    }
}

// ---------------------------------------------------------------------------
extern "C" void kernel_launch(void* const* d_in, const int* in_sizes, int n_in,
                              void* d_out, int out_size, void* d_ws, size_t ws_size,
                              hipStream_t stream) {
    (void)in_sizes; (void)n_in; (void)out_size; (void)ws_size;
    const float* x        = (const float*)d_in[0];
    const int*   position = (const int*)  d_in[1];
    /* d_in[2] = len_region (scalar, compile-time const LENREG) */
    const float* w_qkv    = (const float*)d_in[3];
    const float* b_qkv    = (const float*)d_in[4];
    const float* w_fc     = (const float*)d_in[5];
    const float* b_fc     = (const float*)d_in[6];
    const float* freqs    = (const float*)d_in[7];
    float* out = (float*)d_out;

    const size_t nT    = (size_t)BATCH * NHEADS * SEQ * DHEAD; // 3,145,728 (== B*N*D)
    const size_t nWqkv = (size_t)3 * DMODEL * DMODEL;          // 1,769,472
    const size_t nWfc  = (size_t)DMODEL * DMODEL;              //   589,824

    __bf16* p = (__bf16*)d_ws;
    __bf16* qhi = p; p += nT;   __bf16* qlo = p; p += nT;
    __bf16* khi = p; p += nT;   __bf16* klo = p; p += nT;
    __bf16* vhi = p; p += nT;   __bf16* vlo = p; p += nT;
    __bf16* xhi = p; p += nT;   __bf16* xlo = p; p += nT;
    __bf16* ohi = p; p += nT;   __bf16* olo = p; p += nT;
    __bf16* wqh = p; p += nWqkv; __bf16* wql = p; p += nWqkv;
    __bf16* wfh = p; p += nWfc;  __bf16* wfl = p; p += nWfc;
    // total ~72.4 MB of workspace

    // 0) pack matrix operands into bf16 hi/lo planes (once per launch)
    pack_kernel<<<(int)((nT    + 255) / 256), 256, 0, stream>>>(x,     xhi, xlo, (int)nT);
    pack_kernel<<<(int)((nWqkv + 255) / 256), 256, 0, stream>>>(w_qkv, wqh, wql, (int)nWqkv);
    pack_kernel<<<(int)((nWfc  + 255) / 256), 256, 0, stream>>>(w_fc,  wfh, wfl, (int)nWfc);

    // 1) QKV projection (writes packed Q/K/V directly)
    dim3 g1((3 * DMODEL) / 64, (BATCH * SEQ) / 16);
    qkv_gemm_kernel<<<g1, 32, 0, stream>>>(xhi, xlo, wqh, wql, b_qkv,
                                           qhi, qlo, khi, klo, vhi, vlo);

    // 2) RoPE on core tokens (in place on packed Q/K)
    const int rtot = BATCH * NHEADS * TCORE * (DHEAD / 2);
    rope_kernel<<<(rtot + 255) / 256, 256, 0, stream>>>(qhi, qlo, khi, klo,
                                                        position, freqs);

    // 3) fused flash attention (packed in, packed out)
    dim3 g3(SEQ / 16, NHEADS, BATCH);
    attn_kernel<<<g3, 32, 0, stream>>>(qhi, qlo, khi, klo, vhi, vlo, ohi, olo);

    // 4) output projection -> f32 result
    dim3 g4(DMODEL / 64, (BATCH * SEQ) / 16);
    proj_gemm_kernel<<<g4, 32, 0, stream>>>(ohi, olo, wfh, wfl, b_fc, out);
}
